// Model_91096256348300
// MI455X (gfx1250) — compile-verified, matrix operand order checked by reading, not run on hardware
//
#include <hip/hip_runtime.h>
#include <math.h>
#include <cfloat>

#define CH 64
#define BG 16
#define NN 10000
#define NDEG 16
#define ETOT (BG * NN * NDEG)   // 2,560,000

typedef __attribute__((ext_vector_type(2)))  float    v2f;
typedef __attribute__((ext_vector_type(8)))  float    v8f;
typedef __attribute__((ext_vector_type(16))) _Float16 v16h;

// ---------------- fill helpers ----------------
__global__ void k_fill_f(float* p, float v, int n) {
  int i = blockIdx.x * blockDim.x + threadIdx.x;
  if (i < n) p[i] = v;
}
__global__ void k_fill_i(int* p, int v, int n) {
  int i = blockIdx.x * blockDim.x + threadIdx.x;
  if (i < n) p[i] = v;
}

// ---------------- embedding gather ----------------
__global__ void k_embed(const int* __restrict__ x, const float* __restrict__ emb,
                        float* __restrict__ h, int M) {
  int i = blockIdx.x * blockDim.x + threadIdx.x;
  if (i >= M * CH) return;
  int node = i >> 6, c = i & 63;
  h[i] = emb[x[node] * CH + c];
}

// ---------------- WMMA GEMM: hw[M x 64] = h[M x 64] @ W[64 x 64] ----------------
// W (16 KB) is staged into LDS once per workgroup with CDNA5 async copies
// (GLOBAL_LOAD_ASYNC_TO_LDS_B128, ASYNCcnt). Each wave computes a 32x64 strip
// (two 16-row tiles) so every LDS B-fragment feeds two v_wmma issues.
// Block = 256 threads = 8 waves = 256 rows; M is a multiple of 256
// (160000 / 96000 / 57600).
__global__ __launch_bounds__(256) void k_gemm(const float* __restrict__ h,
                                              const float* __restrict__ W,
                                              float* __restrict__ hw) {
  __shared__ float sW[CH * CH];

#if defined(__AMDGCN__)
  // Async-stage W: 256 threads x 16B x 4 rounds = 16 KB.
  {
    const unsigned ldsBase = (unsigned)(size_t)sW;  // low 32 bits of shared
                                                    // aperture == LDS offset
#pragma unroll
    for (int r = 0; r < 4; ++r) {
      unsigned off = threadIdx.x * 16u + (unsigned)r * 4096u;
      unsigned ldsAddr = ldsBase + off;
      asm volatile("global_load_async_to_lds_b128 %0, %1, %2"
                   :: "v"(ldsAddr), "v"(off), "s"(W) : "memory");
    }
    asm volatile("s_wait_asynccnt 0x0" ::: "memory");
  }
#else
  for (int i = threadIdx.x; i < CH * CH; i += blockDim.x) sW[i] = W[i];
#endif
  __syncthreads();

  const int lane = threadIdx.x & 31;
  const int strip = blockIdx.x * (blockDim.x >> 5) + (threadIdx.x >> 5);
  const int rowBase = strip * 32;
  const int mR0 = rowBase + (lane & 15);

  v8f acc[2][4] = {};

#if defined(__AMDGCN__) && __has_builtin(__builtin_amdgcn_wmma_f32_16x16x4_f32)
  // Full-precision f32 WMMA: K-loop of 16 steps of 16x16x4.
  const int kHi = (lane >> 4) << 1;  // lanes 0-15 -> K+{0,1}; lanes 16-31 -> K+{2,3}
  for (int k0 = 0; k0 < CH; k0 += 4) {
    const int kk = k0 + kHi;
    v2f a0, a1;
    a0.x = h[mR0 * CH + kk];
    a0.y = h[mR0 * CH + kk + 1];
    a1.x = h[(mR0 + 16) * CH + kk];
    a1.y = h[(mR0 + 16) * CH + kk + 1];
#pragma unroll
    for (int t = 0; t < 4; ++t) {
      const int col = t * 16 + (lane & 15);
      v2f b;
      b.x = sW[kk * CH + col];
      b.y = sW[(kk + 1) * CH + col];
      acc[0][t] = __builtin_amdgcn_wmma_f32_16x16x4_f32(
          false, a0, false, b, (short)0, acc[0][t], false, false);
      acc[1][t] = __builtin_amdgcn_wmma_f32_16x16x4_f32(
          false, a1, false, b, (short)0, acc[1][t], false, false);
    }
  }
#else
  // Fallback: f16 inputs, f32 accumulate (codegen-confirmed builtin).
  const int kBase = (lane >> 4) << 3;  // 0 or 8
  for (int k0 = 0; k0 < CH; k0 += 32) {
    v16h a0, a1;
#pragma unroll
    for (int j = 0; j < 16; ++j) {
      int kj = k0 + ((j < 8) ? (kBase + j) : (16 + kBase + (j - 8)));
      a0[j] = (_Float16)h[mR0 * CH + kj];
      a1[j] = (_Float16)h[(mR0 + 16) * CH + kj];
    }
#pragma unroll
    for (int t = 0; t < 4; ++t) {
      const int col = t * 16 + (lane & 15);
      v16h b;
#pragma unroll
      for (int j = 0; j < 16; ++j) {
        int kj = k0 + ((j < 8) ? (kBase + j) : (16 + kBase + (j - 8)));
        b[j] = (_Float16)sW[kj * CH + col];
      }
      acc[0][t] = __builtin_amdgcn_wmma_f32_16x16x32_f16(
          false, a0, false, b, (short)0, acc[0][t], false, false);
      acc[1][t] = __builtin_amdgcn_wmma_f32_16x16x32_f16(
          false, a1, false, b, (short)0, acc[1][t], false, false);
    }
  }
#endif

  // D layout: VGPR v -> (M=base+v) lanes 0-15, (M=base+v+8) lanes 16-31.
#pragma unroll
  for (int rt = 0; rt < 2; ++rt) {
    const int mOut = rowBase + rt * 16 + ((lane >> 4) << 3);
#pragma unroll
    for (int t = 0; t < 4; ++t) {
      const int col = t * 16 + (lane & 15);
#pragma unroll
      for (int v = 0; v < 8; ++v) hw[(mOut + v) * CH + col] = acc[rt][t][v];
    }
  }
}

// ---------------- degree via edge atomics ----------------
__global__ void k_deg(const int* __restrict__ src, const int* __restrict__ dst,
                      const float* __restrict__ ew, float* __restrict__ deg) {
  int e = blockIdx.x * blockDim.x + threadIdx.x;
  if (e >= ETOT) return;
  float w = ew[e];
  if (w != 0.0f) atomicAdd(&deg[dst[e]], w);
}
__global__ void k_dinv(float* deg, int M) {
  int i = blockIdx.x * blockDim.x + threadIdx.x;
  if (i < M) deg[i] = rsqrtf(deg[i] + 1.0f);
}

// ---------------- self-loop term + bias init ----------------
__global__ void k_self(const float* __restrict__ hw, const float* __restrict__ dinv,
                       const float* __restrict__ bias, float* __restrict__ out, int M) {
  int i = blockIdx.x * blockDim.x + threadIdx.x;
  if (i >= M * CH) return;
  int node = i >> 6, c = i & 63;
  float di = dinv[node];
  out[i] = hw[i] * di * di + bias[c];
}

// ---------------- neighborhood aggregation (32 lanes per edge) ----------------
__global__ void k_edge_agg(const int* __restrict__ src, const int* __restrict__ dst,
                           const float* __restrict__ ew, const float* __restrict__ dinv,
                           const float* __restrict__ hw, float* __restrict__ out) {
  int idx = blockIdx.x * blockDim.x + threadIdx.x;
  int e = idx >> 5;
  if (e >= ETOT) return;
  float w = ew[e];
  if (w == 0.0f) return;
  int s = src[e], d = dst[e];
  float coef = dinv[s] * dinv[d] * w;
  int c = (idx & 31) * 2;
  float2 v = *(const float2*)(hw + s * CH + c);
  atomicAdd(out + d * CH + c, v.x * coef);
  atomicAdd(out + d * CH + c + 1, v.y * coef);
}

__global__ void k_relu(float* h, int n) {
  int i = blockIdx.x * blockDim.x + threadIdx.x;
  if (i < n) h[i] = fmaxf(h[i], 0.0f);
}

// ---------------- TopK pooling: scores + order-preserving keys ----------------
__global__ void k_score(const float* __restrict__ h, const float* __restrict__ p,
                        float* __restrict__ score, unsigned* __restrict__ key, int M) {
  int i = blockIdx.x * blockDim.x + threadIdx.x;
  if (i >= M) return;
  float pn = 0.f, s = 0.f;
  for (int c = 0; c < CH; ++c) {
    pn += p[c] * p[c];
    s += h[i * CH + c] * p[c];
  }
  float sc = tanhf(s * rsqrtf(pn));
  score[i] = sc;
  unsigned u = __float_as_uint(sc);
  key[i] = (u & 0x80000000u) ? ~u : (u | 0x80000000u);
}

// Exact k-th-largest per graph via 4-pass radix select; 1 block per graph.
__global__ __launch_bounds__(256) void k_radix(const unsigned* __restrict__ keys,
                                               int n_per, int kSel,
                                               unsigned* __restrict__ thresh,
                                               int* __restrict__ needTies) {
  __shared__ int hist[256];
  __shared__ unsigned s_pref;
  __shared__ int s_rank;
  int g = blockIdx.x;
  const unsigned* kk = keys + (size_t)g * n_per;
  if (threadIdx.x == 0) { s_pref = 0u; s_rank = kSel; }
  __syncthreads();
  unsigned prefmask = 0u;
  for (int shift = 24; shift >= 0; shift -= 8) {
    hist[threadIdx.x] = 0;
    __syncthreads();
    unsigned pref = s_pref;
    for (int i = threadIdx.x; i < n_per; i += 256) {
      unsigned kv = kk[i];
      if ((kv & prefmask) == pref) atomicAdd(&hist[(kv >> shift) & 255u], 1);
    }
    __syncthreads();
    if (threadIdx.x == 0) {
      int r = s_rank, cum = 0, bin = 0;
      for (int b2 = 255; b2 >= 0; --b2) {
        if (cum + hist[b2] >= r) { bin = b2; break; }
        cum += hist[b2];
      }
      s_pref = pref | ((unsigned)bin << shift);
      s_rank = r - cum;
    }
    __syncthreads();
    prefmask |= (0xFFu << shift);
  }
  if (threadIdx.x == 0) { thresh[g] = s_pref; needTies[g] = s_rank; }
}

// Mark kept nodes: all keys > T, plus first needTies keys == T.
__global__ void k_mark(const unsigned* __restrict__ key, const unsigned* __restrict__ thresh,
                       const int* __restrict__ needTies, int* tieCnt, int* outCnt,
                       int* __restrict__ new_idx, int* __restrict__ perm,
                       int n_per, int kSel, int M) {
  int i = blockIdx.x * blockDim.x + threadIdx.x;
  if (i >= M) return;
  int g = i / n_per;
  unsigned kv = key[i], T = thresh[g];
  bool keep = false;
  if (kv > T) keep = true;
  else if (kv == T && atomicAdd(&tieCnt[g], 1) < needTies[g]) keep = true;
  if (keep) {
    int slot = atomicAdd(&outCnt[g], 1);
    int j = g * kSel + slot;
    new_idx[i] = j;
    perm[j] = i;
  }
}

__global__ void k_gather(const float* __restrict__ h, const float* __restrict__ score,
                         const int* __restrict__ perm, float* __restrict__ hn, int Mnew) {
  int i = blockIdx.x * blockDim.x + threadIdx.x;
  if (i >= Mnew * CH) return;
  int j = i >> 6, c = i & 63;
  int old = perm[j];
  hn[i] = h[old * CH + c] * score[old];
}

__global__ void k_renum(int* __restrict__ src, int* __restrict__ dst,
                        float* __restrict__ ew, const int* __restrict__ new_idx) {
  int e = blockIdx.x * blockDim.x + threadIdx.x;
  if (e >= ETOT) return;
  int ns = new_idx[src[e]];
  int nd = new_idx[dst[e]];
  ew[e] *= (ns >= 0 && nd >= 0) ? 1.0f : 0.0f;
  src[e] = ns > 0 ? ns : 0;
  dst[e] = nd > 0 ? nd : 0;
}

// ---------------- global max pool: 1 block/graph, 1 thread/channel ----------------
__global__ void k_gmax(const float* __restrict__ h, float* __restrict__ g, int n_per) {
  int b = blockIdx.x, c = threadIdx.x;
  float m = -FLT_MAX;
  for (int i = 0; i < n_per; ++i) m = fmaxf(m, h[(b * n_per + i) * CH + c]);
  g[b * CH + c] = m;
}

// ---------------- MLP head + log_softmax (single block) ----------------
__global__ __launch_bounds__(128) void k_head(const float* __restrict__ g,
    const float* __restrict__ w0, const float* __restrict__ b0,
    const float* __restrict__ w1, const float* __restrict__ b1,
    const float* __restrict__ w2, const float* __restrict__ b2,
    float* __restrict__ out) {
  __shared__ float t0[BG][CH];
  __shared__ float t1[BG][CH];
  __shared__ float t2[BG][32];
  int tid = threadIdx.x;
  for (int i = tid; i < BG * CH; i += blockDim.x) t0[i >> 6][i & 63] = g[i];
  __syncthreads();
  for (int i = tid; i < BG * CH; i += blockDim.x) {
    int b = i >> 6, c = i & 63;
    float s = b0[c];
    for (int k = 0; k < CH; ++k) s += t0[b][k] * w0[k * CH + c];
    t1[b][c] = fmaxf(s, 0.f);
  }
  __syncthreads();
  for (int i = tid; i < BG * 32; i += blockDim.x) {
    int b = i >> 5, c = i & 31;
    float s = b1[c];
    for (int k = 0; k < CH; ++k) s += t1[b][k] * w1[k * 32 + c];
    t2[b][c] = fmaxf(s, 0.f);
  }
  __syncthreads();
  if (tid < BG) {
    float z0 = b2[0], z1 = b2[1];
    for (int k = 0; k < 32; ++k) { z0 += t2[tid][k] * w2[k * 2]; z1 += t2[tid][k] * w2[k * 2 + 1]; }
    float m = fmaxf(z0, z1);
    float lse = m + logf(expf(z0 - m) + expf(z1 - m));
    out[tid * 2]     = z0 - lse;
    out[tid * 2 + 1] = z1 - lse;
  }
}

// =====================================================================
extern "C" void kernel_launch(void* const* d_in, const int* in_sizes, int n_in,
                              void* d_out, int out_size, void* d_ws, size_t ws_size,
                              hipStream_t stream) {
  (void)in_sizes; (void)n_in; (void)out_size;
  const int*   x      = (const int*)d_in[0];
  const int*   eidx   = (const int*)d_in[1];   // [2, E]
  const float* emb    = (const float*)d_in[3];
  const float* conv_w = (const float*)d_in[4]; // [7,64,64] (in,out)
  const float* conv_b = (const float*)d_in[5];
  const float* pool_p = (const float*)d_in[6];
  const float* fc0_w  = (const float*)d_in[7];
  const float* fc0_b  = (const float*)d_in[8];
  const float* fc1_w  = (const float*)d_in[9];
  const float* fc1_b  = (const float*)d_in[10];
  const float* fc2_w  = (const float*)d_in[11];
  const float* fc2_b  = (const float*)d_in[12];

  const int M0 = BG * NN;  // 160000

  // ---- carve workspace ----
  char* p = (char*)d_ws;
  auto carve = [&](size_t bytes) -> void* {
    void* r = (void*)p;
    p += (bytes + 255) & ~(size_t)255;
    return r;
  };
  float* nbuf[3];
  nbuf[0] = (float*)carve((size_t)M0 * CH * 4);
  nbuf[1] = (float*)carve((size_t)M0 * CH * 4);
  nbuf[2] = (float*)carve((size_t)M0 * CH * 4);
  int*      esrc   = (int*)carve((size_t)ETOT * 4);
  int*      edst   = (int*)carve((size_t)ETOT * 4);
  float*    ew     = (float*)carve((size_t)ETOT * 4);
  float*    deg    = (float*)carve((size_t)M0 * 4);     // becomes dinv in place
  float*    score  = (float*)carve((size_t)M0 * 4);
  unsigned* key    = (unsigned*)carve((size_t)M0 * 4);
  int*      newidx = (int*)carve((size_t)M0 * 4);
  int*      perm   = (int*)carve((size_t)M0 * 4);
  unsigned* thresh = (unsigned*)carve(BG * 4);
  int*      need   = (int*)carve(BG * 4);
  int*      cnts   = (int*)carve(2 * BG * 4);           // tieCnt | outCnt
  float*    gpool  = (float*)carve((size_t)BG * CH * 4);
  if ((size_t)(p - (char*)d_ws) > ws_size) return;      // not enough scratch

  const int T = 256;
  auto nb = [](long long n, int t) { return (unsigned)((n + t - 1) / t); };

  // edges -> scratch (inputs must stay untouched; renumbering mutates in place)
  hipMemcpyAsync(esrc, eidx,        (size_t)ETOT * 4, hipMemcpyDeviceToDevice, stream);
  hipMemcpyAsync(edst, eidx + ETOT, (size_t)ETOT * 4, hipMemcpyDeviceToDevice, stream);
  k_fill_f<<<nb(ETOT, T), T, 0, stream>>>(ew, 1.0f, ETOT);

  // h = embed[x]
  k_embed<<<nb((long long)M0 * CH, T), T, 0, stream>>>(x, emb, nbuf[0], M0);

  int cur = 0;
  int n_per = NN;
  const int ksel[3] = {6000, 3600, 2160};  // ceil(0.6*n) chain

  for (int layer = 0; layer < 7; ++layer) {
    const int M = BG * n_per;
    float* hin  = nbuf[cur];
    float* hw   = nbuf[(cur + 1) % 3];
    float* hout = nbuf[(cur + 2) % 3];

    // degree / dinv
    k_fill_f<<<nb(M, T), T, 0, stream>>>(deg, 0.0f, M);
    k_deg<<<nb(ETOT, T), T, 0, stream>>>(esrc, edst, ew, deg);
    k_dinv<<<nb(M, T), T, 0, stream>>>(deg, M);

    // hw = h @ W  (WMMA; M is a multiple of 256 for every layer: 160000/96000/57600)
    k_gemm<<<M / 256, 256, 0, stream>>>(hin, conv_w + layer * CH * CH, hw);

    // out = hw * dinv^2 + b, then scatter-add neighbor terms, then relu
    k_self<<<nb((long long)M * CH, T), T, 0, stream>>>(hw, deg, conv_b + layer * CH, hout, M);
    k_edge_agg<<<nb((long long)ETOT * 32, T), T, 0, stream>>>(esrc, edst, ew, deg, hw, hout);
    k_relu<<<nb((long long)M * CH, T), T, 0, stream>>>(hout, M * CH);
    cur = (cur + 2) % 3;

    if (layer >= 4) {
      const int j = layer - 4;
      const int k = ksel[j];
      const int Mnew = BG * k;
      float* hn = nbuf[(cur + 1) % 3];  // both other buffers are dead here
      k_score<<<nb(M, T), T, 0, stream>>>(nbuf[cur], pool_p + j * CH, score, key, M);
      k_radix<<<BG, 256, 0, stream>>>(key, n_per, k, thresh, need);
      k_fill_i<<<nb(M, T), T, 0, stream>>>(newidx, -1, M);
      k_fill_i<<<1, 64, 0, stream>>>(cnts, 0, 2 * BG);
      k_mark<<<nb(M, T), T, 0, stream>>>(key, thresh, need, cnts, cnts + BG,
                                         newidx, perm, n_per, k, M);
      k_gather<<<nb((long long)Mnew * CH, T), T, 0, stream>>>(nbuf[cur], score, perm, hn, Mnew);
      k_renum<<<nb(ETOT, T), T, 0, stream>>>(esrc, edst, ew, newidx);
      cur = (cur + 1) % 3;
      n_per = k;
    }
  }

  // global max pool + MLP head
  k_gmax<<<BG, CH, 0, stream>>>(nbuf[cur], gpool, n_per);
  k_head<<<1, 128, 0, stream>>>(gpool, fc0_w, fc0_b, fc1_w, fc1_b, fc2_w, fc2_b,
                                (float*)d_out);
}
